// LSTM_CRF_73297911873606
// MI455X (gfx1250) — matrix operand order repaired and checked by reading, not compiled, not run on hardware
//
#include <hip/hip_runtime.h>
#include <hip/hip_bf16.h>
#include <stdint.h>

// ---------------------------------------------------------------------------
// Problem constants (from the reference)
// ---------------------------------------------------------------------------
#define BB   128     // batch
#define TT   512     // sequence length
#define EE   100     // embedding dim (padded to 128 for WMMA K)
#define EPAD 128
#define HH   128     // hidden
#define G4H  512     // 4*H
#define KK   12      // tags

typedef __attribute__((ext_vector_type(16))) _Float16 v16h;
typedef __attribute__((ext_vector_type(8)))  float    v8f;

__device__ __forceinline__ float sigf(float x) { return 1.0f / (1.0f + __expf(-x)); }

// Async global->LDS copy of one 16B chunk (CDNA5, tracked by ASYNCcnt).
// ldsa = LDS byte offset (VDST), ga = 64-bit global address (VADDR pair).
__device__ __forceinline__ void async_copy_b128(uint32_t ldsa, uint64_t ga) {
    asm volatile("global_load_async_to_lds_b128 %0, %1, off"
                 :: "v"(ldsa), "v"(ga) : "memory");
}
__device__ __forceinline__ void wait_asynccnt0() {
    asm volatile("s_wait_asynccnt 0" ::: "memory");
}

// ---------------------------------------------------------------------------
// 1) Weight prep: pad Wih rows 100->128 in f16, Whh -> f16, bias = bih+bhh
//    grid: 512 blocks x 128 threads
// ---------------------------------------------------------------------------
__global__ void prep_weights_kernel(const float* __restrict__ Wih,
                                    const float* __restrict__ Whh,
                                    const float* __restrict__ bih,
                                    const float* __restrict__ bhh,
                                    _Float16* __restrict__ WihP,   // [512][128]
                                    _Float16* __restrict__ WhhH,   // [512][128]
                                    float* __restrict__ bias) {    // [512]
    int n = blockIdx.x;          // 0..511  (output gate index)
    int k = threadIdx.x;         // 0..127
    WihP[n * EPAD + k] = (k < EE) ? (_Float16)Wih[n * EE + k] : (_Float16)0.0f;
    WhhH[n * HH + k]   = (_Float16)Whh[n * HH + k];
    if (k == 0) bias[n] = bih[n] + bhh[n];
}

// ---------------------------------------------------------------------------
// 2) Embedding gather -> f16, K padded to 128.  X layout: [T][B][128]
//    grid: T*B blocks x 128 threads
// ---------------------------------------------------------------------------
__global__ void gather_embed_kernel(const int* __restrict__ sentences,  // [B][T]
                                    const float* __restrict__ embed,    // [V][100]
                                    _Float16* __restrict__ X) {
    int id = blockIdx.x;             // row = t*B + b
    int t  = id >> 7;                // /128
    int b  = id & 127;
    int k  = threadIdx.x;
    int tok = sentences[b * TT + t];
    X[(size_t)id * EPAD + k] =
        (k < EE) ? (_Float16)embed[(size_t)tok * EE + k] : (_Float16)0.0f;
}

// ---------------------------------------------------------------------------
// 3) Big parallel GEMM: pregate[T*B][512] = X[T*B][128] @ WihP^T
//    block = 256 threads (8 waves).  Block tile: 128(M) x 64(N).
//    grid = (512/64, 65536/128) = (8, 512)
// ---------------------------------------------------------------------------
__global__ void pregate_gemm_kernel(const _Float16* __restrict__ X,     // [M][128]
                                    const _Float16* __restrict__ WihP,  // [512][128]
                                    float* __restrict__ pre) {          // [M][512]
    const int lane = threadIdx.x & 31;
    const int wave = threadIdx.x >> 5;
    const int m0   = blockIdx.y * 128 + wave * 16;
    const int n0b  = blockIdx.x * 64;

    const int arow  = lane & 15;             // A: row within tile
    const int akb   = (lane >> 4) << 4;      // A: lane-half K base (0 or 16)
    const int bcol  = lane & 15;             // B: output column within tile
    const int bkb   = (lane >> 4) << 4;      // B: lane-half K base (0 or 16)

    v8f acc[4];
    #pragma unroll
    for (int nt = 0; nt < 4; ++nt)
        #pragma unroll
        for (int r = 0; r < 8; ++r) acc[nt][r] = 0.0f;

    #pragma unroll
    for (int kc = 0; kc < EPAD; kc += 32) {
        // A fragment 16x32 f16 (contiguous: K = kb+0..15)
        v16h a;
        const _Float16* pa = X + (size_t)(m0 + arow) * EPAD + kc + akb;
        #pragma unroll
        for (int i = 0; i < 16; ++i) a[i] = pa[i];

        #pragma unroll
        for (int nt = 0; nt < 4; ++nt) {
            // B fragment 32x16 f16: lane n holds W[n][kc+bkb .. +15] contiguously
            v16h bf;
            const _Float16* pb = WihP + (size_t)(n0b + nt * 16 + bcol) * EPAD + kc + bkb;
            #pragma unroll
            for (int i = 0; i < 16; ++i) bf[i] = pb[i];
            acc[nt] = __builtin_amdgcn_wmma_f32_16x16x32_f16(
                false, a, false, bf, (short)0, acc[nt], false, false);
        }
    }

    const int mh = lane >> 4;
    const int nl = lane & 15;
    #pragma unroll
    for (int nt = 0; nt < 4; ++nt)
        #pragma unroll
        for (int r = 0; r < 8; ++r)
            pre[(size_t)(m0 + r + 8 * mh) * G4H + n0b + nt * 16 + nl] = acc[nt][r];
}

// ---------------------------------------------------------------------------
// 4) Sequential LSTM recurrence: one persistent workgroup (32 waves, 1 WGP).
//    LDS: Whh f16 (128KB) + h f16 (32KB) + c f32 (64KB) + bias (2KB) = 226KB
//    Whh/c0/bias staged via GLOBAL_LOAD_ASYNC_TO_LDS_B128 (no VGPR round-trip).
//    Wave w owns output rows m0 = (w&7)*16 and hidden cols j0 = (w>>3)*32.
//    Per step/wave: 8 WMMA tiles x 4 K-chunks = 32 v_wmma.
// ---------------------------------------------------------------------------
#define LSTM_SMEM (G4H*HH*2 + BB*HH*2 + BB*HH*4 + G4H*4)

__global__ __launch_bounds__(1024, 1)
void lstm_seq_kernel(const float* __restrict__ pre,       // [T][B][512]
                     const _Float16* __restrict__ WhhG,   // [512][128]
                     const float* __restrict__ bias,      // [512]
                     const float* __restrict__ h0,        // [B][H]
                     const float* __restrict__ c0,        // [B][H]
                     float* __restrict__ hs) {            // [T][B][H]
    extern __shared__ char smem[];
    _Float16* Whh = (_Float16*)smem;                                   // [512][128]
    _Float16* hS  = (_Float16*)(smem + G4H * HH * 2);                  // [128][128]
    float*    cS  = (float*)(smem + G4H * HH * 2 + BB * HH * 2);       // [128][128]
    float*    bS  = (float*)(smem + G4H * HH * 2 + BB * HH * 2 + BB * HH * 4); // [512]

    const int tid = threadIdx.x;

    // ---- async staging: Whh (128KB), c0 (64KB), bias (2KB) straight to LDS ----
    {
        const uint32_t whhL = (uint32_t)(uintptr_t)Whh;
        const uint64_t whhG = (uint64_t)(uintptr_t)WhhG;
        for (int i = tid; i < (G4H * HH * 2) / 16; i += 1024)           // 8192 chunks
            async_copy_b128(whhL + i * 16, whhG + (uint64_t)i * 16);
        const uint32_t cL = (uint32_t)(uintptr_t)cS;
        const uint64_t cG = (uint64_t)(uintptr_t)c0;
        for (int i = tid; i < (BB * HH * 4) / 16; i += 1024)            // 4096 chunks
            async_copy_b128(cL + i * 16, cG + (uint64_t)i * 16);
        const uint32_t bL = (uint32_t)(uintptr_t)bS;
        const uint64_t bG = (uint64_t)(uintptr_t)bias;
        for (int i = tid; i < (G4H * 4) / 16; i += 1024)                // 128 chunks
            async_copy_b128(bL + i * 16, bG + (uint64_t)i * 16);
    }
    // h0 needs f32->f16 conversion: VALU path
    for (int i = tid; i < BB * HH; i += 1024) hS[i] = (_Float16)h0[i];
    wait_asynccnt0();
    __syncthreads();

    const int lane = tid & 31;
    const int wave = tid >> 5;
    const int m0   = (wave & 7) * 16;       // batch-tile
    const int j0   = (wave >> 3) * 32;      // hidden-col block (2 j-tiles)
    const int arow = lane & 15;
    const int akb  = (lane >> 4) << 4;
    const int bcol = lane & 15;
    const int bkb  = (lane >> 4) << 4;
    const int mh   = lane >> 4;
    const int nl   = lane & 15;

    for (int t = 0; t < TT; ++t) {
        // ---- load A fragments of current h (reads of hS) ----
        v16h A[4];
        #pragma unroll
        for (int kc = 0; kc < 4; ++kc) {
            const _Float16* pa = hS + (m0 + arow) * HH + kc * 32 + akb;
            #pragma unroll
            for (int i = 0; i < 16; ++i) A[kc][i] = pa[i];
        }
        __syncthreads();   // all reads of hS done before anyone rewrites it

        // prefetch next timestep's pre-gates while matrix cores run
        if (t + 1 < TT) {
            const float* pf = pre + ((size_t)(t + 1) * BB + m0 + 8 * mh) * G4H + j0 + nl;
            __builtin_prefetch(pf, 0, 1);
        }

        #pragma unroll
        for (int jj = 0; jj < 2; ++jj) {
            v8f acc[4];   // i, f, g, o tiles for this j-tile
            #pragma unroll
            for (int g = 0; g < 4; ++g) {
                const int n0 = g * HH + j0 + jj * 16;
                // accumulator init: pregate (x@Wih^T) + (bih+bhh)
                const float* pp = pre + ((size_t)t * BB + m0 + 8 * mh) * G4H + n0 + nl;
                const float  bv = bS[n0 + nl];
                v8f d;
                #pragma unroll
                for (int r = 0; r < 8; ++r) d[r] = pp[(size_t)r * G4H] + bv;

                #pragma unroll
                for (int kc = 0; kc < 4; ++kc) {
                    v16h bf;
                    const _Float16* pb = Whh + (n0 + bcol) * HH + kc * 32 + bkb;
                    #pragma unroll
                    for (int i = 0; i < 16; ++i) bf[i] = pb[i];
                    d = __builtin_amdgcn_wmma_f32_16x16x32_f16(
                        false, A[kc], false, bf, (short)0, d, false, false);
                }
                acc[g] = d;
            }
            // ---- elementwise cell update (wave-exclusive region of cS/hS) ----
            const int j = j0 + jj * 16 + nl;
            #pragma unroll
            for (int r = 0; r < 8; ++r) {
                const int b = m0 + 8 * mh + r;
                float iv = sigf(acc[0][r]);
                float fv = sigf(acc[1][r]);
                float gv = tanhf(acc[2][r]);
                float ov = sigf(acc[3][r]);
                float c  = fv * cS[b * HH + j] + iv * gv;
                cS[b * HH + j] = c;
                float h = ov * tanhf(c);
                hS[b * HH + j] = (_Float16)h;
                hs[((size_t)t * BB + b) * HH + j] = h;
            }
        }
        __syncthreads();   // new h visible to all waves before next step's reads
    }
}

// ---------------------------------------------------------------------------
// 5) Emissions: emit[t][b][k] = hs[t][b][:] . Wtag[k][:] + btag[k]
//    grid: T blocks x 128 threads (thread = batch element)
// ---------------------------------------------------------------------------
__global__ void emissions_kernel(const float* __restrict__ hs,
                                 const float* __restrict__ Wtag,   // [12][128]
                                 const float* __restrict__ btag,
                                 float* __restrict__ emit) {       // [T][B][12]
    __shared__ float Wt[KK * HH];
    __shared__ float bt[KK];
    const int t = blockIdx.x;
    for (int i = threadIdx.x; i < KK * HH; i += blockDim.x) Wt[i] = Wtag[i];
    if (threadIdx.x < KK) bt[threadIdx.x] = btag[threadIdx.x];
    __syncthreads();

    const int b = threadIdx.x;
    const float* h = hs + ((size_t)t * BB + b) * HH;
    float acc[KK];
    #pragma unroll
    for (int k = 0; k < KK; ++k) acc[k] = bt[k];
    for (int x = 0; x < HH; ++x) {
        const float hv = h[x];
        #pragma unroll
        for (int k = 0; k < KK; ++k) acc[k] += hv * Wt[k * HH + x];
    }
    #pragma unroll
    for (int k = 0; k < KK; ++k) emit[((size_t)t * BB + b) * KK + k] = acc[k];
}

// ---------------------------------------------------------------------------
// 6) Viterbi forward + backtrace.  grid: B blocks x 32 threads.
//    out[0..127] = scores; out[128 + b*T + t] = path tag (as float).
// ---------------------------------------------------------------------------
__global__ void viterbi_kernel(const float* __restrict__ emit,    // [T][B][12]
                               const float* __restrict__ trans,   // [12][12]
                               unsigned char* __restrict__ bp,    // [T-1][B][12]
                               float* __restrict__ out) {
    __shared__ float tr[KK * KK];
    __shared__ float prev[KK];
    __shared__ float curr[KK];
    const int b = blockIdx.x;
    const int j = threadIdx.x;

    for (int i = threadIdx.x; i < KK * KK; i += 32) tr[i] = trans[i];
    if (j < KK) prev[j] = emit[(size_t)b * KK + j];
    __syncthreads();

    for (int t = 1; t < TT; ++t) {
        if (j < KK) {
            float best = prev[0] + tr[0 * KK + j];
            int   bi   = 0;
            #pragma unroll
            for (int i = 1; i < KK; ++i) {
                float v = prev[i] + tr[i * KK + j];
                if (v > best) { best = v; bi = i; }
            }
            bp[((size_t)(t - 1) * BB + b) * KK + j] = (unsigned char)bi;
            curr[j] = emit[((size_t)t * BB + b) * KK + j] + best;
        }
        __syncthreads();
        if (j < KK) prev[j] = curr[j];
        __syncthreads();
    }

    if (j == 0) {
        float best = prev[0];
        int   tag  = 0;
        #pragma unroll
        for (int i = 1; i < KK; ++i)
            if (prev[i] > best) { best = prev[i]; tag = i; }
        out[b] = best;
        out[BB + (size_t)b * TT + (TT - 1)] = (float)tag;
        for (int t = TT - 2; t >= 0; --t) {
            tag = bp[((size_t)t * BB + b) * KK + tag];
            out[BB + (size_t)b * TT + t] = (float)tag;
        }
    }
}

// ---------------------------------------------------------------------------
// Host orchestration
// ---------------------------------------------------------------------------
extern "C" void kernel_launch(void* const* d_in, const int* in_sizes, int n_in,
                              void* d_out, int out_size, void* d_ws, size_t ws_size,
                              hipStream_t stream) {
    const int*   sentences = (const int*)  d_in[0];
    const float* h0        = (const float*)d_in[1];
    const float* c0        = (const float*)d_in[2];
    const float* embed     = (const float*)d_in[3];
    const float* Wih       = (const float*)d_in[4];
    const float* Whh       = (const float*)d_in[5];
    const float* bih       = (const float*)d_in[6];
    const float* bhh       = (const float*)d_in[7];
    const float* Wtag      = (const float*)d_in[8];
    const float* btag      = (const float*)d_in[9];
    const float* trans     = (const float*)d_in[10];
    float* out = (float*)d_out;

    // carve workspace
    char* w = (char*)d_ws;
    auto carve = [&](size_t bytes) -> char* {
        char* p = w;
        w += (bytes + 255) & ~(size_t)255;
        return p;
    };
    _Float16* X     = (_Float16*)carve((size_t)TT * BB * EPAD * 2);   // 16.8 MB
    _Float16* WihP  = (_Float16*)carve((size_t)G4H * EPAD * 2);
    _Float16* WhhH  = (_Float16*)carve((size_t)G4H * HH * 2);
    float*    bias  = (float*)   carve((size_t)G4H * 4);
    float*    pre   = (float*)   carve((size_t)TT * BB * G4H * 4);    // 134 MB
    float*    hs    = (float*)   carve((size_t)TT * BB * HH * 4);     // 33.5 MB
    float*    emit  = (float*)   carve((size_t)TT * BB * KK * 4);
    unsigned char* bp = (unsigned char*)carve((size_t)(TT - 1) * BB * KK);

    // allow >64KB dynamic LDS for the persistent LSTM kernel
    hipFuncSetAttribute((const void*)lstm_seq_kernel,
                        hipFuncAttributeMaxDynamicSharedMemorySize, LSTM_SMEM);

    prep_weights_kernel<<<G4H, EPAD, 0, stream>>>(Wih, Whh, bih, bhh, WihP, WhhH, bias);
    gather_embed_kernel<<<TT * BB, EPAD, 0, stream>>>(sentences, embed, X);
    pregate_gemm_kernel<<<dim3(G4H / 64, (TT * BB) / 128), 256, 0, stream>>>(X, WihP, pre);
    lstm_seq_kernel<<<1, 1024, LSTM_SMEM, stream>>>(pre, WhhH, bias, h0, c0, hs);
    emissions_kernel<<<TT, BB, 0, stream>>>(hs, Wtag, btag, emit);
    viterbi_kernel<<<BB, 32, 0, stream>>>(emit, trans, bp, out);
}
